// Eff_GAT_18674517803417
// MI455X (gfx1250) — matrix operand order repaired and checked by reading, not compiled, not run on hardware
//
#include <hip/hip_runtime.h>
#include <math.h>

#define NVN 20000
#define NEE 320000
#define DIMF 1152
#define HIDF 256

typedef __attribute__((ext_vector_type(16))) __bf16 v16bf;
typedef __attribute__((ext_vector_type(8)))  float  v8f;

union FragAB { v16bf v; uint4 u[2]; };

__device__ __forceinline__ float gelu_f(float x) {
  return 0.5f * x * (1.0f + erff(x * 0.70710678118654752f));
}

__device__ __forceinline__ unsigned short f2bf(float f) {
  unsigned int u = __float_as_uint(f);
  if ((u & 0x7fffffffu) > 0x7f800000u) return 0x7fc0;   // NaN
  u += 0x7fffu + ((u >> 16) & 1u);                      // round-to-nearest-even
  return (unsigned short)(u >> 16);
}

__device__ __forceinline__ void atomicMaxF(float* addr, float val) {
  if (val >= 0.0f) atomicMax((int*)addr, __float_as_int(val));
  else             atomicMin((unsigned int*)addr, __float_as_uint(val));
}

// ---------------------------------------------------------------------------
// WMMA GEMM:  C[M,N] = act( A[M,K](bf16,row-major) * Bt[N,K](bf16)^T + bias )
// wave32; each wave owns a 32x32 output region (2x2 WMMA tiles) so each K-step
// issues 8 b128 fragment loads for 4 v_wmma ops. Out-of-range output columns
// are handled by CLAMPING the B row pointer (a B column only feeds the same D
// column), so the inner loop has no EXEC masking and no zero-fill; only the
// epilogue stores are guarded. Fragment layouts per CDNA5 ISA 7.12.2.
// Requires: K % 32 == 0, M % 32 == 0.
// ---------------------------------------------------------------------------
__global__ void __launch_bounds__(256)
wmma_gemm_bf16(const unsigned short* __restrict__ A,
               const unsigned short* __restrict__ Bt,
               const float* __restrict__ bias,
               float* __restrict__ C,
               int M, int K, int N, int doGelu)
{
  const int lane = threadIdx.x & 31;
  const int wave = threadIdx.x >> 5;
  const int wm = wave >> 2;                   // 0..1
  const int wn = wave & 3;                    // 0..3
  const int mbase = blockIdx.y * 64 + wm * 32;
  const int nbase = blockIdx.x * 128 + wn * 32;
  if (mbase >= M || nbase >= N) return;       // wave-uniform: EXEC stays full
  const int hi  = lane >> 4;                  // K-half selector
  const int lid = lane & 15;

  const int c0 = nbase + lid;
  const int c1 = nbase + 16 + lid;
  const int c0c = (c0 < N) ? c0 : (N - 1);    // clamp: loads in-bounds,
  const int c1c = (c1 < N) ? c1 : (N - 1);    // garbage cols never stored

  const unsigned short* a0 = A  + (size_t)(mbase + lid) * K;
  const unsigned short* a1 = A  + (size_t)(mbase + 16 + lid) * K;
  const unsigned short* b0 = Bt + (size_t)c0c * K;
  const unsigned short* b1 = Bt + (size_t)c1c * K;

  v8f acc00 = {}, acc01 = {}, acc10 = {}, acc11 = {};
  for (int k0 = 0; k0 < K; k0 += 32) {
    if (k0 + 128 < K) {
      __builtin_prefetch(a0 + k0 + 128, 0, 3);   // global_prefetch_b8 (near)
      __builtin_prefetch(a1 + k0 + 128, 0, 3);
    }
    FragAB fa0, fa1, fb0, fb1;
    // A 16x32 bf16: lanes 0-15 K={0..7,16..23}, lanes 16-31 K={8..15,24..31}
    fa0.u[0] = *(const uint4*)(a0 + k0 + hi * 8);
    fa0.u[1] = *(const uint4*)(a0 + k0 + 16 + hi * 8);
    fa1.u[0] = *(const uint4*)(a1 + k0 + hi * 8);
    fa1.u[1] = *(const uint4*)(a1 + k0 + 16 + hi * 8);
    // B 32x16 bf16 (transposed storage): lane = N col; lanes 0-15 K={0..7,16..23}, lanes 16-31 K={8..15,24..31}
    fb0.u[0] = *(const uint4*)(b0 + k0 + hi * 8);
    fb0.u[1] = *(const uint4*)(b0 + k0 + 16 + hi * 8);
    fb1.u[0] = *(const uint4*)(b1 + k0 + hi * 8);
    fb1.u[1] = *(const uint4*)(b1 + k0 + 16 + hi * 8);
    acc00 = __builtin_amdgcn_wmma_f32_16x16x32_bf16(false, fa0.v, false, fb0.v,
                                                    (short)0, acc00, false, false);
    acc01 = __builtin_amdgcn_wmma_f32_16x16x32_bf16(false, fa0.v, false, fb1.v,
                                                    (short)0, acc01, false, false);
    acc10 = __builtin_amdgcn_wmma_f32_16x16x32_bf16(false, fa1.v, false, fb0.v,
                                                    (short)0, acc10, false, false);
    acc11 = __builtin_amdgcn_wmma_f32_16x16x32_bf16(false, fa1.v, false, fb1.v,
                                                    (short)0, acc11, false, false);
  }

  auto emit = [&](v8f acc, int mt, int col, bool ok) {
    if (!ok) return;
    const float bv = bias ? bias[col] : 0.0f;
#pragma unroll
    for (int r = 0; r < 8; ++r) {             // D: VGPR r -> M = mt + hi*8 + r
      float vv = acc[r] + bv;
      if (doGelu) vv = gelu_f(vv);
      C[(size_t)(mt + hi * 8 + r) * N + col] = vv;
    }
  };
  emit(acc00, mbase,      c0, c0 < N);
  emit(acc01, mbase,      c1, c1 < N);
  emit(acc10, mbase + 16, c0, c0 < N);
  emit(acc11, mbase + 16, c1, c1 < N);
}

// ---------------------------------------------------------------------------
// Conversions
// ---------------------------------------------------------------------------
__global__ void cvt_bf16_k(const float* __restrict__ in,
                           unsigned short* __restrict__ out, int n) {
  int i = blockIdx.x * blockDim.x + threadIdx.x;
  if (i < n) out[i] = f2bf(in[i]);
}

// W[K,N] fp32 row-major -> Bt[N,K] bf16 (transpose + convert)
__global__ void cvt_t_bf16_k(const float* __restrict__ W,
                             unsigned short* __restrict__ Bt, int K, int N) {
  int i = blockIdx.x * blockDim.x + threadIdx.x;
  if (i >= K * N) return;
  int k = i / N, n = i - k * N;
  Bt[(size_t)n * K + k] = f2bf(W[i]);
}

__global__ void act_store_k(const float* __restrict__ in,
                            float* __restrict__ outf,
                            unsigned short* __restrict__ outb,
                            int n, int doGelu) {
  int i = blockIdx.x * blockDim.x + threadIdx.x;
  if (i >= n) return;
  float v = in[i];
  if (doGelu) v = gelu_f(v);
  outf[i] = v;
  outb[i] = f2bf(v);
}

__global__ void resid_cvt_k(const float* __restrict__ x,
                            const float* __restrict__ c,
                            unsigned short* __restrict__ outb, int n) {
  int i = blockIdx.x * blockDim.x + threadIdx.x;
  if (i < n) outb[i] = f2bf(x[i] + c[i]);
}

// ---------------------------------------------------------------------------
// Feature assembly: concat(patch, posMLP(xy), temb[time]) per node
// ---------------------------------------------------------------------------
__global__ void build_combined_k(const float* __restrict__ xy,
                                 const int* __restrict__ timev,
                                 const float* __restrict__ patch,
                                 const float* __restrict__ temb,
                                 const float* __restrict__ pw1, const float* __restrict__ pb1,
                                 const float* __restrict__ pw2, const float* __restrict__ pb2,
                                 float* __restrict__ comb, int n) {
  int i = blockIdx.x * blockDim.x + threadIdx.x;
  if (i >= n) return;
  const float* p = patch + (size_t)i * 1088;
  float* c = comb + (size_t)i * DIMF;
  for (int j = 0; j < 1088; ++j) c[j] = p[j];
  float x0 = xy[i * 2], x1 = xy[i * 2 + 1];
  float h[16];
#pragma unroll
  for (int j = 0; j < 16; ++j) h[j] = gelu_f(x0 * pw1[j] + x1 * pw1[16 + j] + pb1[j]);
  for (int j = 0; j < 32; ++j) {
    float a = pb2[j];
#pragma unroll
    for (int t = 0; t < 16; ++t) a += h[t] * pw2[t * 32 + j];
    c[1088 + j] = a;
  }
  int tt = timev[i];
  for (int j = 0; j < 32; ++j) c[1120 + j] = temb[tt * 32 + j];
}

// ---------------------------------------------------------------------------
// Edge / attention pipeline
// ---------------------------------------------------------------------------
__global__ void init_heads_k(float* __restrict__ nmax, float* __restrict__ nden, int n) {
  int i = blockIdx.x * blockDim.x + threadIdx.x;
  if (i >= n) return;
  nmax[i] = -INFINITY;
  nden[i] = 0.0f;
}

__global__ void nmax_fix_k(float* __restrict__ m, int n) {
  int i = blockIdx.x * blockDim.x + threadIdx.x;
  if (i >= n) return;
  if (m[i] < -3.0e38f) m[i] = 0.0f;   // segment_max of empty segment -> 0
}

__global__ void edge_scores_k(const float* __restrict__ q, const float* __restrict__ k,
                              const int* __restrict__ src, const int* __restrict__ dst,
                              float* __restrict__ alpha, float* __restrict__ nmax,
                              int E, int dh, int od, float scale) {
  int idx = blockIdx.x * blockDim.x + threadIdx.x;
  if (idx >= E * 8) return;
  int e = idx >> 3, h = idx & 7;
  int s = src[e], d = dst[e];
  const float* qp = q + (size_t)d * od + h * dh;
  const float* kp = k + (size_t)s * od + h * dh;
  float acc = 0.0f;
  for (int j = 0; j < dh; ++j) acc += qp[j] * kp[j];
  acc *= scale;
  alpha[idx] = acc;
  atomicMaxF(&nmax[d * 8 + h], acc);
}

__global__ void edge_exp_k(float* __restrict__ alpha, const int* __restrict__ dst,
                           const float* __restrict__ nmax, float* __restrict__ nden, int E) {
  int idx = blockIdx.x * blockDim.x + threadIdx.x;
  if (idx >= E * 8) return;
  int e = idx >> 3, h = idx & 7;
  int d = dst[e];
  float ev = expf(alpha[idx] - nmax[d * 8 + h]);
  alpha[idx] = ev;
  atomicAdd(&nden[d * 8 + h], ev);
}

__global__ void edge_agg_k(float* __restrict__ alpha, const float* __restrict__ v,
                           const int* __restrict__ src, const int* __restrict__ dst,
                           const float* __restrict__ nden, float* __restrict__ xo,
                           int E, int dh, int od) {
  int idx = blockIdx.x * blockDim.x + threadIdx.x;
  if (idx >= E * 8) return;
  int e = idx >> 3, h = idx & 7;
  int s = src[e], d = dst[e];
  float a = alpha[idx] / nden[d * 8 + h];
  alpha[idx] = a;                                   // final attention output
  const float* vp = v + (size_t)s * od + h * dh;
  float* op = xo + (size_t)d * od + h * dh;
  for (int j = 0; j < dh; ++j) atomicAdd(&op[j], a * vp[j]);
}

// ---------------------------------------------------------------------------
extern "C" void kernel_launch(void* const* d_in, const int* in_sizes, int n_in,
                              void* d_out, int out_size, void* d_ws, size_t ws_size,
                              hipStream_t stream) {
  (void)in_sizes; (void)n_in; (void)out_size; (void)ws_size;
  const float* xy    = (const float*)d_in[0];
  const int*   timev = (const int*)d_in[1];
  const float* patch = (const float*)d_in[2];
  const int*   eidx  = (const int*)d_in[3];
  const float* temb  = (const float*)d_in[5];
  const float* pw1   = (const float*)d_in[6];
  const float* pb1   = (const float*)d_in[7];
  const float* pw2   = (const float*)d_in[8];
  const float* pb2   = (const float*)d_in[9];
  const float* mw1   = (const float*)d_in[10];
  const float* mb1   = (const float*)d_in[11];
  const float* mw2   = (const float*)d_in[12];
  const float* mb2   = (const float*)d_in[13];
  const float* W0    = (const float*)d_in[14];
  const float* b0    = (const float*)d_in[15];
  const float* Wm    = (const float*)d_in[16];
  const float* bm    = (const float*)d_in[17];
  const float* W3    = (const float*)d_in[18];
  const float* b3    = (const float*)d_in[19];
  const float* fw1   = (const float*)d_in[20];
  const float* fb1   = (const float*)d_in[21];
  const float* fw2   = (const float*)d_in[22];
  const float* fb2   = (const float*)d_in[23];

  float* out = (float*)d_out;
  const int* src = eidx;
  const int* dst = eidx + NEE;

  // bump-allocate scratch
  char* wsp = (char*)d_ws;
  auto alloc = [&](size_t bytes) -> char* {
    char* p = wsp; wsp += (bytes + 255) & ~(size_t)255; return p;
  };
  float* comb            = (float*)alloc((size_t)NVN * DIMF * 4);
  unsigned short* combb  = (unsigned short*)alloc((size_t)NVN * DIMF * 2);
  float* xf              = (float*)alloc((size_t)NVN * DIMF * 4);
  unsigned short* xb     = (unsigned short*)alloc((size_t)NVN * DIMF * 2);
  float* q               = (float*)alloc((size_t)NVN * DIMF * 4);
  float* kbuf            = (float*)alloc((size_t)NVN * DIMF * 4);
  float* vbuf            = (float*)alloc((size_t)NVN * DIMF * 4);
  float* xo              = (float*)alloc((size_t)NVN * DIMF * 4);
  float* h128            = (float*)alloc((size_t)NVN * 128 * 4);
  unsigned short* h128b  = (unsigned short*)alloc((size_t)NVN * 128 * 2);
  float* nmax            = (float*)alloc((size_t)NVN * 8 * 4);
  float* nden            = (float*)alloc((size_t)NVN * 8 * 4);
  unsigned short* mw1t   = (unsigned short*)alloc((size_t)DIMF * 128 * 2);
  unsigned short* mw2t   = (unsigned short*)alloc((size_t)128 * DIMF * 2);
  unsigned short* W0t    = (unsigned short*)alloc((size_t)4 * DIMF * HIDF * 2);
  unsigned short* Wmt    = (unsigned short*)alloc((size_t)8 * HIDF * HIDF * 2);
  unsigned short* W3t    = (unsigned short*)alloc((size_t)4 * HIDF * DIMF * 2);
  unsigned short* fw1t   = (unsigned short*)alloc((size_t)DIMF * 32 * 2);
  unsigned short* fw2t   = (unsigned short*)alloc((size_t)32 * 2 * 2);

  auto cvtT = [&](const float* W, unsigned short* Bt, int K, int N) {
    int tot = K * N;
    cvt_t_bf16_k<<<(tot + 255) / 256, 256, 0, stream>>>(W, Bt, K, N);
  };
  auto gemm = [&](const unsigned short* A, const unsigned short* Bt, const float* bias,
                  float* C, int M, int K, int N, int act) {
    dim3 g((unsigned)((N + 127) / 128), (unsigned)((M + 63) / 64));
    wmma_gemm_bf16<<<g, 256, 0, stream>>>(A, Bt, bias, C, M, K, N, act);
  };

  // --- weight conversion (fp32 -> transposed bf16), once per launch ---
  cvtT(mw1, mw1t, DIMF, 128);
  cvtT(mw2, mw2t, 128, DIMF);
  for (int m = 0; m < 4; ++m)
    cvtT(W0 + (size_t)m * DIMF * HIDF, W0t + (size_t)m * HIDF * DIMF, DIMF, HIDF);
  for (int i = 0; i < 8; ++i)
    cvtT(Wm + (size_t)i * HIDF * HIDF, Wmt + (size_t)i * HIDF * HIDF, HIDF, HIDF);
  for (int m = 0; m < 4; ++m)
    cvtT(W3 + (size_t)m * HIDF * DIMF, W3t + (size_t)m * DIMF * HIDF, HIDF, DIMF);
  cvtT(fw1, fw1t, DIMF, 32);
  cvtT(fw2, fw2t, 32, 2);

  // --- feature assembly + mixer MLP ---
  build_combined_k<<<(NVN + 255) / 256, 256, 0, stream>>>(
      xy, timev, patch, temb, pw1, pb1, pw2, pb2, comb, NVN);
  cvt_bf16_k<<<(NVN * DIMF + 255) / 256, 256, 0, stream>>>(comb, combb, NVN * DIMF);
  gemm(combb, mw1t, mb1, h128, NVN, DIMF, 128, 1);
  cvt_bf16_k<<<(NVN * 128 + 255) / 256, 256, 0, stream>>>(h128, h128b, NVN * 128);
  gemm(h128b, mw2t, mb2, comb, NVN, 128, DIMF, 0);            // post-mixer "combined"
  cvt_bf16_k<<<(NVN * DIMF + 255) / 256, 256, 0, stream>>>(comb, xb, NVN * DIMF);

  // --- 4 TransformerConv layers ---
  struct Cfg { const unsigned short* Wt; const float* b; int ind, od, dh; };
  Cfg L[4] = {
    { W0t,                               b0,            DIMF, HIDF, 32  },
    { Wmt,                               bm,            HIDF, HIDF, 32  },
    { Wmt + (size_t)4 * HIDF * HIDF,     bm + 4 * HIDF, HIDF, HIDF, 32  },
    { W3t,                               b3,            HIDF, DIMF, 144 },
  };
  for (int l = 0; l < 4; ++l) {
    const int ind = L[l].ind, od = L[l].od, dh = L[l].dh;
    const unsigned short* Wq = L[l].Wt;
    const unsigned short* Wk = Wq + (size_t)od * ind;
    const unsigned short* Wv = Wk + (size_t)od * ind;
    const unsigned short* Ws = Wv + (size_t)od * ind;
    const float* bq = L[l].b;
    const float* bk = bq + od;
    const float* bv = bk + od;
    const float* bs = bv + od;

    gemm(xb, Wq, bq, q,    NVN, ind, od, 0);
    gemm(xb, Wk, bk, kbuf, NVN, ind, od, 0);
    gemm(xb, Wv, bv, vbuf, NVN, ind, od, 0);
    gemm(xb, Ws, bs, xo,   NVN, ind, od, 0);   // skip term: aggregation adds on top

    init_heads_k<<<(NVN * 8 + 255) / 256, 256, 0, stream>>>(nmax, nden, NVN * 8);
    float* alpha = out + 2 * NVN + (size_t)l * NEE * 8;
    const float scale = 1.0f / sqrtf((float)dh);
    edge_scores_k<<<(NEE * 8 + 255) / 256, 256, 0, stream>>>(
        q, kbuf, src, dst, alpha, nmax, NEE, dh, od, scale);
    nmax_fix_k<<<(NVN * 8 + 255) / 256, 256, 0, stream>>>(nmax, NVN * 8);
    edge_exp_k<<<(NEE * 8 + 255) / 256, 256, 0, stream>>>(alpha, dst, nmax, nden, NEE);
    edge_agg_k<<<(NEE * 8 + 255) / 256, 256, 0, stream>>>(
        alpha, vbuf, src, dst, nden, xo, NEE, dh, od);

    act_store_k<<<(NVN * od + 255) / 256, 256, 0, stream>>>(
        xo, xf, xb, NVN * od, (l < 3) ? 1 : 0);
  }

  // --- final head: gelu((x + combined) @ fw1 + fb1) @ fw2 + fb2 ---
  resid_cvt_k<<<(NVN * DIMF + 255) / 256, 256, 0, stream>>>(xf, comb, xb, NVN * DIMF);
  gemm(xb, fw1t, fb1, h128, NVN, DIMF, 32, 1);
  cvt_bf16_k<<<(NVN * 32 + 255) / 256, 256, 0, stream>>>(h128, h128b, NVN * 32);
  gemm(h128b, fw2t, fb2, out, NVN, 32, 2, 0);
}
